// CRF_69123203662217
// MI455X (gfx1250) — matrix operand order repaired and checked by reading, not compiled, not run on hardware
//
#include <hip/hip_runtime.h>
#include <math.h>

typedef __attribute__((ext_vector_type(16))) _Float16 v16h;
typedef __attribute__((ext_vector_type(8)))  float    v8f;

#define NB   64      // batch
#define NT   128     // seq len
#define NTAG 10000
#define EMB  128
#define NK   500     // neg samples
#define SC   512     // padded K (32 col-tiles of 16, 16 k-chunks of 32)
#define NEG_INF (-1.0e30f)

// ---------------- Kernel A: We = emb @ W_w^T + b  (10000 x 128) ----------------
// Block computes a 16x128 tile; the 16 emb rows staged in LDS, each thread streams
// its W_w row exactly once (L2 traffic: 625 * 64KB = 40MB instead of 640MB).
__global__ void we_kernel(const float* __restrict__ emb, const float* __restrict__ Ww,
                          const float* __restrict__ Wb, float* __restrict__ We) {
    __shared__ float eb[16][EMB];
    int c = threadIdx.x;
    int r0 = blockIdx.x * 16;
    #pragma unroll
    for (int r = 0; r < 16; ++r) eb[r][c] = emb[(size_t)(r0 + r) * EMB + c];
    __syncthreads();
    float acc[16];
    #pragma unroll
    for (int r = 0; r < 16; ++r) acc[r] = 0.f;
    for (int k = 0; k < EMB; ++k) {
        float w = Ww[c * EMB + k];
        #pragma unroll
        for (int r = 0; r < 16; ++r) acc[r] += eb[r][k] * w;
    }
    float bias = Wb[c];
    #pragma unroll
    for (int r = 0; r < 16; ++r) We[(size_t)(r0 + r) * EMB + c] = acc[r] + bias;
}

// ---------------- Kernel B: TR[i][j] = A[neg_i,neg_j]*relu(We[neg_i].emb[neg_j]) ------
__global__ void trans_kernel(const float* __restrict__ We, const float* __restrict__ emb,
                             const float* __restrict__ A, const int* __restrict__ neg,
                             float* __restrict__ TR) {
    __shared__ float Wt[16][EMB];
    __shared__ float Et[16][EMB];
    int ti = blockIdx.x, tj = blockIdx.y, tid = threadIdx.x;
    for (int idx = tid; idx < 16 * EMB; idx += 256) {
        int row = idx >> 7, k = idx & 127;
        int i = ti * 16 + row; int gi = (i < NK) ? neg[i] : 0;
        Wt[row][k] = We[gi * EMB + k];
        int j = tj * 16 + row; int gj = (j < NK) ? neg[j] : 0;
        Et[row][k] = emb[gj * EMB + k];
    }
    __syncthreads();
    int li = tid >> 4, lj = tid & 15;
    int i = ti * 16 + li, j = tj * 16 + lj;
    float d = 0.f;
    #pragma unroll 8
    for (int k = 0; k < EMB; ++k) d += Wt[li][k] * Et[lj][k];
    float outv = NEG_INF;
    if (i < NK && j < NK)
        outv = A[(size_t)neg[i] * NTAG + neg[j]] * fmaxf(d, 0.f);
    TR[i * SC + j] = outv;
}

// ---------------- Kernel B2: per-column max of TR ----------------
__global__ void cmax_kernel(const float* __restrict__ TR, float* __restrict__ cmax) {
    int j = threadIdx.x;
    float m = NEG_INF;
    for (int i = 0; i < SC; ++i) m = fmaxf(m, TR[i * SC + j]);
    cmax[j] = m;
}

// ---------------- Kernel B3: P = exp(TR - cmax), packed in f16 WMMA B-fragment layout ----
// B (32K x 16N) per wave: lane n=l&15; lanes 0-15 hold K=0..15 (VGPR v -> K=2v,2v+1),
// lanes 16-31 hold K=16..31.  Frag f = cb*16+kc; storage [f][lane][16 halves].
__global__ void pack_kernel(const float* __restrict__ TR, const float* __restrict__ cmax,
                            _Float16* __restrict__ Pfrag) {
    int g = blockIdx.x * 256 + threadIdx.x;   // 0..16383
    int f = g >> 5, lane = g & 31;
    int cb = f >> 4, kc = f & 15;
    int n = lane & 15, lh = lane >> 4;
    int j = cb * 16 + n;
    float cm = cmax[j];
    #pragma unroll
    for (int h = 0; h < 16; ++h) {
        int v = h >> 1, p = h & 1;
        int kl = lh * 16 + v * 2 + p;          // K within chunk
        int i = kc * 32 + kl;
        float x = expf(TR[i * SC + j] - cm);   // in [0,1]: safe in f16
        Pfrag[((size_t)f * 32 + lane) * 16 + h] = (_Float16)x;
    }
}

// ---------------- Kernel C: em[t][b][j] = emissions[b][t][neg[j]] (pad -> -inf) --------
__global__ void gather_kernel(const float* __restrict__ emis, const int* __restrict__ neg,
                              float* __restrict__ em) {
    int t = blockIdx.x, b = blockIdx.y, j = threadIdx.x;
    float v = NEG_INF;
    if (j < NK) v = emis[((size_t)(b * NT + t)) * NTAG + neg[j]];
    em[((size_t)(t * NB + b)) * SC + j] = v;
}

// ---------------- Kernel D: numerator[b] (tag-path score) ----------------
__global__ void numer_kernel(const float* __restrict__ emis, const int* __restrict__ tags,
                             const float* __restrict__ We, const float* __restrict__ emb,
                             const float* __restrict__ A, float* __restrict__ numer) {
    __shared__ float red[NT];
    int b = blockIdx.x, t = threadIdx.x;
    float val;
    if (t == 0) {
        val = emis[(size_t)(b * NT) * NTAG + tags[b * NT]];
    } else {
        int tp = tags[b * NT + t - 1], tc = tags[b * NT + t];
        float d = 0.f;
        #pragma unroll 8
        for (int k = 0; k < EMB; ++k) d += We[tp * EMB + k] * emb[tc * EMB + k];
        d = fmaxf(d, 0.f);
        val = A[(size_t)tp * NTAG + tc] * d + emis[((size_t)(b * NT + t)) * NTAG + tc];
    }
    red[t] = val;
    __syncthreads();
    for (int s = 64; s; s >>= 1) { if (t < s) red[t] += red[t + s]; __syncthreads(); }
    if (t == 0) numer[b] = red[0];
}

// ---------------- Kernel E: the scan. One persistent workgroup, 16 waves. --------------
// Per step: nxt[b,j] = log( sum_i exp(score[b,i]-m_b) * P[i,j] ) + m_b + cmax[j] + em[t,b,j]
// A-side s lives in LDS in f16 A-fragment layout (16x32 per (rb,kc)):
//   lane l: M = l&15; lanes<16 hold K 0-15, lanes>=16 hold K 16-31
__global__ __launch_bounds__(512)
void scan_kernel(const _Float16* __restrict__ Pfrag, const float* __restrict__ em,
                 const float* __restrict__ cmax_g, float* __restrict__ scoreOut) {
    __shared__ float    cmaxl[SC];
    __shared__ float    mprev[NB];
    __shared__ float    mnew[NB];
    __shared__ float    wmax[NB * 4];
    __shared__ _Float16 sfrag[4 * 16 * 32 * 16];   // [rb][kc][lane][16] = 64KB

    const int tid    = threadIdx.x;
    const int lane   = tid & 31;
    const int w      = tid >> 5;
    const int rb     = w >> 2;          // row block 0..3   (rows of 16 batches)
    const int widx   = w & 3;           // which wave within row block
    const int n      = lane & 15;       // output column within tile
    const int halfid = lane >> 4;       // C/D: M = r + 8*halfid

    cmaxl[tid] = cmax_g[tid];
    __syncthreads();

    for (int t = 0; t < NT; ++t) {
        // Prefetch this step's emission slice (wave footprint: 16 rows x 512B =
        // 64 x 128B lines; one address per lane x2 -> global_prefetch_b8) so the
        // epilogue's em loads hit near caches after the GEMM covers the latency.
        {
            const char* base =
                (const char*)&em[((size_t)(t * NB + rb * 16)) * SC + widx * 128];
            int l0 = lane, l1 = lane + 32;
            __builtin_prefetch(base + (size_t)(l0 >> 2) * (SC * 4) + (l0 & 3) * 128, 0, 0);
            __builtin_prefetch(base + (size_t)(l1 >> 2) * (SC * 4) + (l1 & 3) * 128, 0, 0);
        }

        v8f acc[8];
        if (t == 0) {
            // score0 = em_neg[0]  (pads already -inf)
            #pragma unroll
            for (int ci = 0; ci < 8; ++ci) {
                int j = (widx * 8 + ci) * 16 + n;
                #pragma unroll
                for (int r = 0; r < 8; ++r) {
                    int brow = rb * 16 + r + 8 * halfid;
                    acc[ci][r] = em[(size_t)brow * SC + j];   // t==0 slice
                }
            }
        } else {
            #pragma unroll
            for (int ci = 0; ci < 8; ++ci) acc[ci] = (v8f){0,0,0,0,0,0,0,0};
            // GEMM: s(64x512,f16 in LDS) @ P(512x512,f16 in L2) -> f32.
            // Preload all 8 B-fragments of the k-chunk into independent regs so the
            // scheduler emits one wide load clause and staggers s_wait_loadcnt
            // instead of a wait(0) per WMMA.
            for (int kc = 0; kc < 16; ++kc) {
                v16h a = *(const v16h*)&sfrag[(((size_t)rb * 16 + kc) * 32 + lane) * 16];
                const _Float16* pb =
                    &Pfrag[(((size_t)(widx * 8) * 16 + kc) * 32 + lane) * 16];
                v16h bb[8];
                #pragma unroll
                for (int ci = 0; ci < 8; ++ci)
                    bb[ci] = *(const v16h*)(pb + (size_t)ci * (16 * 32 * 16));
                #pragma unroll
                for (int ci = 0; ci < 8; ++ci)
                    acc[ci] = __builtin_amdgcn_wmma_f32_16x16x32_f16(
                        false, a, false, bb[ci], (short)0, acc[ci], false, false);
            }
            // log + shifts + emission
            #pragma unroll
            for (int ci = 0; ci < 8; ++ci) {
                int j = (widx * 8 + ci) * 16 + n;
                #pragma unroll
                for (int r = 0; r < 8; ++r) {
                    int brow = rb * 16 + r + 8 * halfid;
                    float v = acc[ci][r];
                    float nv = NEG_INF;
                    if (j < NK)
                        nv = logf(fmaxf(v, 1e-37f)) + mprev[brow] + cmaxl[j]
                           + em[((size_t)(t * NB + brow)) * SC + j];
                    acc[ci][r] = nv;
                }
            }
        }
        // ---- per-row max (over this wave's 128 columns, then across 4 waves) ----
        float rmax[8];
        #pragma unroll
        for (int r = 0; r < 8; ++r) {
            float m = NEG_INF;
            #pragma unroll
            for (int ci = 0; ci < 8; ++ci) m = fmaxf(m, acc[ci][r]);
            #pragma unroll
            for (int msk = 1; msk < 16; msk <<= 1)
                m = fmaxf(m, __shfl_xor(m, msk, 16));   // reduce over n within half
            rmax[r] = m;
        }
        if (n == 0) {
            #pragma unroll
            for (int r = 0; r < 8; ++r)
                wmax[(rb * 16 + r + 8 * halfid) * 4 + widx] = rmax[r];
        }
        __syncthreads();
        if (tid < NB) {
            float m = fmaxf(fmaxf(wmax[tid * 4 + 0], wmax[tid * 4 + 1]),
                            fmaxf(wmax[tid * 4 + 2], wmax[tid * 4 + 3]));
            mnew[tid] = m;
        }
        __syncthreads();

        if (t == NT - 1) {
            // final scores out (pads store -inf; finish kernel reads j<500 anyway)
            #pragma unroll
            for (int ci = 0; ci < 8; ++ci) {
                int j = (widx * 8 + ci) * 16 + n;
                #pragma unroll
                for (int r = 0; r < 8; ++r) {
                    int brow = rb * 16 + r + 8 * halfid;
                    scoreOut[(size_t)brow * SC + j] = acc[ci][r];
                }
            }
        } else {
            // s' = exp(score - m) packed into A-fragment layout in LDS
            #pragma unroll
            for (int ci = 0; ci < 8; ++ci) {
                int j = (widx * 8 + ci) * 16 + n;
                int kc = j >> 5, kl = j & 31;
                int kh = (kl >> 4) & 1;
                int v  = (kl & 15) >> 1;
                int p  = kl & 1;
                #pragma unroll
                for (int r = 0; r < 8; ++r) {
                    int m = r + 8 * halfid;
                    int brow = rb * 16 + m;
                    float s = expf(acc[ci][r] - mnew[brow]);  // 0 for pads (-inf)
                    sfrag[(((size_t)rb * 16 + kc) * 32 + (m + 16 * kh)) * 16 + (v * 2 + p)]
                        = (_Float16)s;
                }
            }
        }
        __syncthreads();
        if (tid < NB) mprev[tid] = mnew[tid];
        __syncthreads();
    }
}

// ---------------- Kernel F: denominator + final scalar ----------------
__global__ void finish_kernel(const float* __restrict__ score, const float* __restrict__ numer,
                              float* __restrict__ out) {
    __shared__ float llh[NB];
    int tid = threadIdx.x, w = tid >> 5, lane = tid & 31;
    for (int b = w; b < NB; b += 16) {
        float m = NEG_INF;
        for (int j = lane; j < NK; j += 32) m = fmaxf(m, score[b * SC + j]);
        for (int s = 16; s; s >>= 1) m = fmaxf(m, __shfl_xor(m, s, 32));
        float nb = numer[b];
        m = fmaxf(m, nb);
        float sum = 0.f;
        for (int j = lane; j < NK; j += 32) sum += expf(score[b * SC + j] - m);
        for (int s = 16; s; s >>= 1) sum += __shfl_xor(sum, s, 32);
        if (lane == 0) {
            sum += expf(nb - m);
            float denom = m + logf(sum) + logf((float)NTAG / (float)(NK + 1));
            llh[b] = nb - denom;
        }
    }
    __syncthreads();
    if (tid == 0) {
        float tot = 0.f;
        for (int b = 0; b < NB; ++b) tot += llh[b];
        out[0] = tot / (float)(NB * NT);   // mask all-true -> mf.sum() = 8192
    }
}

extern "C" void kernel_launch(void* const* d_in, const int* in_sizes, int n_in,
                              void* d_out, int out_size, void* d_ws, size_t ws_size,
                              hipStream_t stream) {
    const float* emis = (const float*)d_in[0];
    const int*   tags = (const int*)d_in[1];
    const float* emb  = (const float*)d_in[2];
    const float* A    = (const float*)d_in[3];
    // d_in[4] = mask (all true in this problem; folded into constants)
    const int*   neg  = (const int*)d_in[5];
    const float* Ww   = (const float*)d_in[6];
    const float* Wb   = (const float*)d_in[7];
    float* out = (float*)d_out;

    char* ws = (char*)d_ws;
    float*    We    = (float*)(ws);                    //  5,120,000 B
    float*    TR    = (float*)(ws + 5120000);          //  1,048,576 B
    float*    cmax  = (float*)(ws + 6168576);          //      2,048 B
    _Float16* Pfrag = (_Float16*)(ws + 6170624);       //    524,288 B
    float*    em    = (float*)(ws + 6694912);          // 16,777,216 B
    float*    numer = (float*)(ws + 23472128);         //        256 B
    float*    score = (float*)(ws + 23472384);         //    131,072 B

    we_kernel    <<<NTAG / 16, EMB, 0, stream>>>(emb, Ww, Wb, We);
    trans_kernel <<<dim3(SC/16, SC/16), 256, 0, stream>>>(We, emb, A, neg, TR);
    cmax_kernel  <<<1, SC, 0, stream>>>(TR, cmax);
    pack_kernel  <<<64, 256, 0, stream>>>(TR, cmax, Pfrag);
    gather_kernel<<<dim3(NT, NB), SC, 0, stream>>>(emis, neg, em);
    numer_kernel <<<NB, NT, 0, stream>>>(emis, tags, We, emb, A, numer);
    scan_kernel  <<<1, 512, 0, stream>>>(Pfrag, em, cmax, score);
    finish_kernel<<<1, 512, 0, stream>>>(score, numer, out);
}
